// NetBinary_8907762172387
// MI455X (gfx1250) — compile-verified, matrix operand order checked by reading, not run on hardware
//
#include <hip/hip_runtime.h>
#include <hip/hip_bf16.h>

#define EPS_BN 1e-5f

typedef int v8i __attribute__((ext_vector_type(8)));

// K index inside a 64-wide block for 8-bit A/B WMMA fragments (ISA 7.12.2):
// VGPR0: K=base+0..3, VGPR1: K=base+4..7, VGPR2/3: +16, VGPR4..7: +32,
// base = 8 for lanes 16..31.
__device__ __forceinline__ int kfrag_idx(int v, int b, int lane) {
    return 32 * (v >> 2) + 16 * ((v >> 1) & 1) + ((lane & 16) ? 8 : 0) + 4 * (v & 1) + b;
}

__global__ void k_zero(float* __restrict__ p, long long n) {
    long long i = (long long)blockIdx.x * blockDim.x + threadIdx.x;
    long long s = (long long)gridDim.x * blockDim.x;
    for (; i < n; i += s) p[i] = 0.0f;
}

// Per-channel sum / sumsq over (N, HW). x layout [N, C, HW]. stats[2c]=sum, [2c+1]=sumsq.
__global__ void k_bn_stats(const float* __restrict__ x, float* __restrict__ stats,
                           int C, int HW, long long per_c) {
    int c = blockIdx.y;
    float s = 0.0f, s2 = 0.0f;
    long long CHW = (long long)C * HW;
    for (long long i = (long long)blockIdx.x * blockDim.x + threadIdx.x; i < per_c;
         i += (long long)gridDim.x * blockDim.x) {
        long long n = i / HW, p = i % HW;
        float v = x[n * CHW + (long long)c * HW + p];
        s += v; s2 += v * v;
    }
    for (int off = 16; off > 0; off >>= 1) {
        s  += __shfl_down(s,  off, 32);
        s2 += __shfl_down(s2, off, 32);
    }
    if ((threadIdx.x & 31) == 0) {
        atomicAdd(&stats[2 * c],     s);
        atomicAdd(&stats[2 * c + 1], s2);
    }
}

// BatchNorm (training-mode, biased var) + binarize -> int8 sign (+1/-1/0).
__global__ void k_bn_sign(const float* __restrict__ x, const float* __restrict__ stats,
                          const float* __restrict__ gamma, const float* __restrict__ beta,
                          signed char* __restrict__ out, int C, int HW,
                          long long total, float inv_cnt) {
    long long i = (long long)blockIdx.x * blockDim.x + threadIdx.x;
    long long s = (long long)gridDim.x * blockDim.x;
    for (; i < total; i += s) {
        int c = (int)((i / HW) % C);
        float m  = stats[2 * c] * inv_cnt;
        float vv = stats[2 * c + 1] * inv_cnt - m * m;
        float xn = (x[i] - m) * rsqrtf(vv + EPS_BN) * gamma[c] + beta[c];
        out[i] = (xn > 0.0f) ? 1 : ((xn < 0.0f) ? -1 : 0);
    }
}

__global__ void k_sign(const float* __restrict__ w, signed char* __restrict__ out, long long n) {
    long long i = (long long)blockIdx.x * blockDim.x + threadIdx.x;
    long long s = (long long)gridDim.x * blockDim.x;
    for (; i < n; i += s) {
        float v = w[i];
        out[i] = (v > 0.0f) ? 1 : ((v < 0.0f) ? -1 : 0);
    }
}

// Fragment layouts (dword granularity), lane-contiguous so GEMM loads are B128:
//   dword index = ((tile*KB + kb)*32 + lane)*8 + v

// Pre-pack binarized weights [Cout, Kreal] into B fragments (64x16 int8 per block).
__global__ void k_pack_b(const signed char* __restrict__ sw, int* __restrict__ bfrag,
                         int Cout, int Kreal, int KB, long long total) {
    long long i = (long long)blockIdx.x * blockDim.x + threadIdx.x;
    long long s = (long long)gridDim.x * blockDim.x;
    for (; i < total; i += s) {
        int v    = (int)(i & 7);
        int lane = (int)((i >> 3) & 31);
        long long t = i >> 8;
        int kb = (int)(t % KB);
        int tn = (int)(t / KB);
        int cout = tn * 16 + (lane & 15);
        unsigned dw = 0;
        #pragma unroll
        for (int b = 0; b < 4; ++b) {
            int k = kb * 64 + kfrag_idx(v, b, lane);
            int val = 0;
            if (cout < Cout && k < Kreal)
                val = sw[(long long)cout * Kreal + k];
            dw |= ((unsigned)(val & 0xFF)) << (8 * b);
        }
        bfrag[i] = (int)dw;
    }
}

// Explicit im2col of sign activations into A fragments (16x64 int8 per block),
// chunked over M tiles: tile index in buffer is (tileM - tileM0).
__global__ void k_pack_a(const signed char* __restrict__ sa, int* __restrict__ afrag,
                         int Cin, int H, int W, int Ho, int Wo, int KH, int KW,
                         int Kreal, int KB, int M, int tileM0, long long total) {
    long long i = (long long)blockIdx.x * blockDim.x + threadIdx.x;
    long long s = (long long)gridDim.x * blockDim.x;
    const int HoWo = Ho * Wo;
    const int KHW  = KH * KW;
    for (; i < total; i += s) {
        int v    = (int)(i & 7);
        int lane = (int)((i >> 3) & 31);
        long long t = i >> 8;
        int kb = (int)(t % KB);
        int tl = (int)(t / KB);
        int m  = (tileM0 + tl) * 16 + (lane & 15);
        unsigned dw = 0;
        if (m < M) {
            int n  = m / HoWo;
            int r  = m % HoWo;
            int oh = r / Wo;
            int ow = r % Wo;
            long long rowbase = ((long long)n * Cin) * H * W;
            #pragma unroll
            for (int b = 0; b < 4; ++b) {
                int k = kb * 64 + kfrag_idx(v, b, lane);
                int val = 0;
                if (k < Kreal) {
                    int cin = k / KHW;
                    int rr  = k % KHW;
                    int kh  = rr / KW;
                    int kw  = rr % KW;
                    val = sa[rowbase + ((long long)cin * H + (oh + kh)) * W + (ow + kw)];
                }
                dw |= ((unsigned)(val & 0xFF)) << (8 * b);
            }
        }
        afrag[i] = (int)dw;
    }
}

// Binarized GEMM with 2x2 register blocking: one wave computes a 32(M) x 32(Cout)
// block (four 16x16 WMMA tiles). Inner loop: 8x global_load_b128 + 4x
// V_WMMA_I32_16X16X64_IU8 -> 2x the MACs per byte of L2 fragment traffic.
__global__ void __launch_bounds__(32)
k_gemm_wmma(const v8i* __restrict__ afrag, const v8i* __restrict__ bfrag,
            int* __restrict__ outI,
            int Cout, int Ho, int Wo, int KB, int M, int tileM0,
            int tcM, int tilesN) {
    const int lane = threadIdx.x;
    // clamp edge tiles to last valid tile: redundant identical compute, keeps
    // loads in range and EXEC all-ones at the WMMAs; stores are bounds-guarded.
    int tl0 = blockIdx.x * 2;
    int tl1 = tl0 + 1; if (tl1 > tcM - 1) tl1 = tcM - 1;
    int tn0 = blockIdx.y * 2;
    int tn1 = tn0 + 1; if (tn1 > tilesN - 1) tn1 = tilesN - 1;

    const int HoWo = Ho * Wo;

    v8i acc00 = {}, acc01 = {}, acc10 = {}, acc11 = {};
    long long a0off = ((long long)tl0 * KB) * 32 + lane;
    long long a1off = ((long long)tl1 * KB) * 32 + lane;
    long long b0off = ((long long)tn0 * KB) * 32 + lane;
    long long b1off = ((long long)tn1 * KB) * 32 + lane;

    for (int kb = 0; kb < KB; ++kb) {
        long long ko = (long long)kb * 32;
        v8i a0 = afrag[a0off + ko];
        v8i a1 = afrag[a1off + ko];
        v8i b0 = bfrag[b0off + ko];
        v8i b1 = bfrag[b1off + ko];
        acc00 = __builtin_amdgcn_wmma_i32_16x16x64_iu8(true, a0, true, b0, acc00, false, false);
        acc01 = __builtin_amdgcn_wmma_i32_16x16x64_iu8(true, a0, true, b1, acc01, false, false);
        acc10 = __builtin_amdgcn_wmma_i32_16x16x64_iu8(true, a1, true, b0, acc10, false, false);
        acc11 = __builtin_amdgcn_wmma_i32_16x16x64_iu8(true, a1, true, b1, acc11, false, false);
    }

    // C/D layout: lanes0-15: VGPR r = (M=r, N=lane); lanes16-31: (M=8+r, N=lane-16)
    const int laneN = lane & 15;
    const int rbase = (lane & 16) ? 8 : 0;
    const int tiles[4][2] = {{tl0, tn0}, {tl0, tn1}, {tl1, tn0}, {tl1, tn1}};
    v8i accs[4] = {acc00, acc01, acc10, acc11};
    #pragma unroll
    for (int j = 0; j < 4; ++j) {
        const int cout  = tiles[j][1] * 16 + laneN;
        const int mbase = (tileM0 + tiles[j][0]) * 16 + rbase;
        #pragma unroll
        for (int r = 0; r < 8; ++r) {
            int mm = mbase + r;
            if (mm < M && cout < Cout) {
                int n2 = mm / HoWo;
                int rr = mm % HoWo;
                long long idx = (((long long)n2 * Cout + cout) * Ho + rr / Wo) * Wo + (rr % Wo);
                outI[idx] = accs[j][r];
            }
        }
    }
}

// 2x2 max-pool on integer conv output, then PReLU -> f32
__global__ void k_pool_prelu(const int* __restrict__ in, float* __restrict__ out,
                             const float* __restrict__ alpha,
                             int C, int Ho, int Wo, int Hp, int Wp, long long total) {
    long long i = (long long)blockIdx.x * blockDim.x + threadIdx.x;
    long long s = (long long)gridDim.x * blockDim.x;
    float a = alpha[0];
    for (; i < total; i += s) {
        long long t = i;
        int pw = (int)(t % Wp); t /= Wp;
        int ph = (int)(t % Hp); t /= Hp;
        int c  = (int)(t % C);
        int n  = (int)(t / C);
        long long base = (((long long)n * C + c) * Ho + 2 * ph) * Wo + 2 * pw;
        int v0 = in[base], v1 = in[base + 1], v2 = in[base + Wo], v3 = in[base + Wo + 1];
        int mx = v0 > v1 ? v0 : v1;
        int my = v2 > v3 ? v2 : v3;
        int m  = mx > my ? mx : my;
        float f = (float)m;
        out[i] = (f >= 0.0f) ? f : a * f;
    }
}

__global__ void k_prelu_i2f(const int* __restrict__ in, float* __restrict__ out,
                            const float* __restrict__ alpha, long long total) {
    long long i = (long long)blockIdx.x * blockDim.x + threadIdx.x;
    long long s = (long long)gridDim.x * blockDim.x;
    float a = alpha[0];
    for (; i < total; i += s) {
        float f = (float)in[i];
        out[i] = (f >= 0.0f) ? f : a * f;
    }
}

__global__ void k_prelu_scale(const int* __restrict__ in, float* __restrict__ out,
                              const float* __restrict__ alpha, const float* __restrict__ scale,
                              long long total) {
    long long i = (long long)blockIdx.x * blockDim.x + threadIdx.x;
    long long s = (long long)gridDim.x * blockDim.x;
    float a = alpha[0], sc = scale[0];
    for (; i < total; i += s) {
        float f = (float)in[i];
        f = (f >= 0.0f) ? f : a * f;
        out[i] = f * sc;
    }
}

static inline int clampBlocks(long long work, int tpb, int maxb) {
    long long b = (work + tpb - 1) / tpb;
    if (b > maxb) b = maxb;
    if (b < 1) b = 1;
    return (int)b;
}

extern "C" void kernel_launch(void* const* d_in, const int* in_sizes, int n_in,
                              void* d_out, int out_size, void* d_ws, size_t ws_size,
                              hipStream_t stream) {
    (void)in_sizes; (void)n_in; (void)out_size; (void)ws_size;
    const float* x     = (const float*)d_in[0];
    const float* cg[3] = {(const float*)d_in[1], (const float*)d_in[5], (const float*)d_in[9]};
    const float* cb[3] = {(const float*)d_in[2], (const float*)d_in[6], (const float*)d_in[10]};
    const float* cw[3] = {(const float*)d_in[3], (const float*)d_in[7], (const float*)d_in[11]};
    const float* cp[3] = {(const float*)d_in[4], (const float*)d_in[8], (const float*)d_in[12]};
    const float* fg[2] = {(const float*)d_in[13], (const float*)d_in[17]};
    const float* fb[2] = {(const float*)d_in[14], (const float*)d_in[18]};
    const float* fw[2] = {(const float*)d_in[15], (const float*)d_in[19]};
    const float* fp[2] = {(const float*)d_in[16], (const float*)d_in[20]};
    const float* scale = (const float*)d_in[21];
    float* out = (float*)d_out;

    // ---- workspace bump allocator ----
    char* ws = (char*)d_ws;
    size_t off = 0;
    auto walloc = [&](size_t bytes) -> void* {
        void* p = (void*)(ws + off);
        off += (bytes + 255) & ~(size_t)255;
        return p;
    };
    float*       stats = (float*)walloc(2ull * 18432 * sizeof(float));  // max C = 18432
    signed char* signS = (signed char*)walloc(16ull << 20);             // max 15.8 MB (L1 input)
    signed char* sw    = (signed char*)walloc(19ull << 20);             // max 18.9 MB (fw0)
    int*         bfrag = (int*)walloc(19ull << 20);                     // packed B fragments
    int*         afrag = (int*)walloc(48ull << 20);                     // packed A fragments (chunked)
    int*         convI = (int*)walloc(253ull << 20);                    // max 252 MB (L0 conv out)
    float*       actA  = (float*)walloc(64ull << 20);                   // max 63 MB (L0 pooled)
    float*       actB  = (float*)walloc(26ull << 20);                   // max 25.7 MB (L1 pooled)

    const int N = 128;
    const int TPB = 256;
    const int CHUNK_TILES = 2048;   // afrag chunk: <= 2048*KB*1024 bytes (max 42.5 MB at KB=36)

    auto bn_pass = [&](const float* xin, const float* g, const float* b,
                       int C, int HW, signed char* so) {
        long long per_c = (long long)N * HW;
        long long total = (long long)C * per_c;
        k_zero<<<clampBlocks(2LL * C, TPB, 256), TPB, 0, stream>>>(stats, 2LL * C);
        dim3 gs((unsigned)clampBlocks(per_c, TPB, 512), (unsigned)C);
        k_bn_stats<<<gs, TPB, 0, stream>>>(xin, stats, C, HW, per_c);
        k_bn_sign<<<clampBlocks(total, TPB, 4096), TPB, 0, stream>>>(
            xin, stats, g, b, so, C, HW, total, 1.0f / (float)per_c);
    };

    auto gemm_pass = [&](const signed char* sa, const float* wf,
                         int Cin, int H, int W, int Cout, int KH, int KW, int Ho, int Wo) {
        int Kreal  = Cin * KH * KW;
        int KB     = (Kreal + 63) / 64;
        int tilesN = (Cout + 15) / 16;
        int M      = N * Ho * Wo;
        int tilesM = (M + 15) / 16;

        long long wn = (long long)Cout * Kreal;
        k_sign<<<clampBlocks(wn, TPB, 4096), TPB, 0, stream>>>(wf, sw, wn);
        long long pbd = (long long)tilesN * KB * 256;
        k_pack_b<<<clampBlocks(pbd, TPB, 4096), TPB, 0, stream>>>(sw, bfrag, Cout, Kreal, KB, pbd);

        for (int t0 = 0; t0 < tilesM; t0 += CHUNK_TILES) {
            int tc = tilesM - t0;
            if (tc > CHUNK_TILES) tc = CHUNK_TILES;
            long long pad = (long long)tc * KB * 256;
            k_pack_a<<<clampBlocks(pad, TPB, 4096), TPB, 0, stream>>>(
                sa, afrag, Cin, H, W, Ho, Wo, KH, KW, Kreal, KB, M, t0, pad);
            dim3 gg((unsigned)((tc + 1) / 2), (unsigned)((tilesN + 1) / 2));
            k_gemm_wmma<<<gg, 32, 0, stream>>>((const v8i*)afrag, (const v8i*)bfrag, convI,
                                               Cout, Ho, Wo, KB, M, t0, tc, tilesN);
        }
    };

    auto pool_pass = [&](float* dst, const float* alpha, int C, int Ho, int Wo) {
        int Hp = Ho / 2, Wp = Wo / 2;
        long long total = (long long)N * C * Hp * Wp;
        k_pool_prelu<<<clampBlocks(total, TPB, 4096), TPB, 0, stream>>>(
            convI, dst, alpha, C, Ho, Wo, Hp, Wp, total);
    };

    // ---- conv block 0: [128,3,64,64] -> conv 3->128 -> pool -> [128,128,31,31] ----
    bn_pass(x, cg[0], cb[0], 3, 64 * 64, signS);
    gemm_pass(signS, cw[0], 3, 64, 64, 128, 3, 3, 62, 62);
    pool_pass(actA, cp[0], 128, 62, 62);

    // ---- conv block 1: -> conv 128->256 -> pool -> [128,256,14,14] ----
    bn_pass(actA, cg[1], cb[1], 128, 31 * 31, signS);
    gemm_pass(signS, cw[1], 128, 31, 31, 256, 3, 3, 29, 29);
    pool_pass(actB, cp[1], 256, 29, 29);

    // ---- conv block 2: -> conv 256->512 -> pool -> [128,512,6,6] ----
    bn_pass(actB, cg[2], cb[2], 256, 14 * 14, signS);
    gemm_pass(signS, cw[2], 256, 14, 14, 512, 3, 3, 12, 12);
    pool_pass(actA, cp[2], 512, 12, 12);

    // ---- FC0: flatten [128,18432] -> 1024 ----
    bn_pass(actA, fg[0], fb[0], 18432, 1, signS);
    gemm_pass(signS, fw[0], 18432, 1, 1, 1024, 1, 1, 1, 1);
    {
        long long total = (long long)N * 1024;
        k_prelu_i2f<<<clampBlocks(total, TPB, 1024), TPB, 0, stream>>>(convI, actB, fp[0], total);
    }

    // ---- FC1: [128,1024] -> 10, then scale ----
    bn_pass(actB, fg[1], fb[1], 1024, 1, signS);
    gemm_pass(signS, fw[1], 1024, 1, 1, 10, 1, 1, 1, 1);
    {
        long long total = (long long)N * 10;
        k_prelu_scale<<<clampBlocks(total, TPB, 64), TPB, 0, stream>>>(convI, out, fp[1], scale, total);
    }
}
